// SparseMoERewardModel_54606214201798
// MI455X (gfx1250) — matrix operand order repaired and checked by reading, not compiled, not run on hardware
//
#include <hip/hip_runtime.h>
#include <hip/hip_bf16.h>

// ---------------------------------------------------------------------------
// Problem constants (from the reference)
// ---------------------------------------------------------------------------
#define B_     2048
#define NA_    8
#define D_     512
#define E_     4
#define H_     8
#define DH_    64
#define FFN_   2048
#define HEAD_  512
#define BINS_  101
#define NTOK   (B_ * NA_)     // 16384 tokens
#define FLATD  (NA_ * D_)     // 4096
#define LN_EPS 1e-5f

typedef __attribute__((ext_vector_type(16))) _Float16 hvec16;
typedef __attribute__((ext_vector_type(8)))  _Float16 hvec8;
typedef __attribute__((ext_vector_type(8)))  float    fvec8;

// ---------------------------------------------------------------------------
// WMMA fragment loaders (CDNA5 ISA 7.12.2 layouts, wave32)
// A 16x32 f16: lane<16 -> row=lane, halves[0..7]=K0..7, halves[8..15]=K16..23
//              lane>=16 -> row=lane-16, halves=K8..15 / K24..31
// B 32x16 f16: lane<16 -> col=lane, halves[0..7]=K0..7, halves[8..15]=K16..23
//              lane>=16 -> col=lane-16, halves=K8..15 / K24..31
// C/D 16x16 f32: vgpr r -> M=r (lanes 0-15, N=lane) / M=8+r (lanes 16-31)
// ---------------------------------------------------------------------------
__device__ inline hvec16 load_fragA(const _Float16* tile, int ld, int lane) {
  int r = lane & 15;
  int g = (lane >> 4) << 3;                 // 0 or 8
  const _Float16* row = tile + r * ld;
  hvec8 lo = *(const hvec8*)(row + g);
  hvec8 hi = *(const hvec8*)(row + g + 16);
  return __builtin_shufflevector(lo, hi, 0,1,2,3,4,5,6,7,8,9,10,11,12,13,14,15);
}

__device__ inline hvec16 load_fragB(const _Float16* tile, int ld, int lane) {
  int r = lane & 15;
  int g = (lane >> 4) << 3;                 // 0 or 8
  const _Float16* row = tile + r * ld;
  hvec8 lo = *(const hvec8*)(row + g);
  hvec8 hi = *(const hvec8*)(row + g + 16);
  return __builtin_shufflevector(lo, hi, 0,1,2,3,4,5,6,7,8,9,10,11,12,13,14,15);
}

// ---------------------------------------------------------------------------
// Tiled WMMA GEMM:  C[MxN] = A[MxK](f16,row-major) * BT[NxK]^T (+bias)(+relu)
// Block tile 128x128, K-step 32, 256 threads = 8 waves (4 M-waves x 2 N-waves),
// each wave owns a 32x64 region = 2x4 wmma accumulators.
// Staging: gfx1250 async Global->LDS DMA (ASYNCcnt) with double-buffered LDS.
// Per thread per tile: 4 x global_load_async_to_lds_b128 (A:512 + B:512 chunks
// of 8 halves across 256 threads) -> ASYNCcnt += 4 per wave per tile.
// ---------------------------------------------------------------------------
#define GBM 128
#define GBN 128
#define GBK 32
#define LDSROW 40   // 32 + 8 pad halves (80B row) -> conflict-free ds_load_b128

__global__ __launch_bounds__(256) void gemm_f16_wmma_kernel(
    const _Float16* __restrict__ A, const _Float16* __restrict__ BT,
    const float* __restrict__ bias, float* __restrict__ C32,
    _Float16* __restrict__ C16, int M, int N, int Kd, int relu) {
  __shared__ __align__(16) _Float16 As[2][GBM * LDSROW];
  __shared__ __align__(16) _Float16 Bs[2][GBN * LDSROW];

  const int tid  = threadIdx.x;
  const int lane = tid & 31;
  const int wave = tid >> 5;
  const int wm   = wave & 3;      // 0..3 -> 32-row slice
  const int wn   = wave >> 2;     // 0..1 -> 64-col slice
  const int m0   = blockIdx.x * GBM;
  const int n0   = blockIdx.y * GBN;

  // ---- precompute per-thread async staging addresses (4 chunks/thread) ----
  // chunk c in [0,1024): c<512 -> A tile (row=c/4, seg=(c%4)*8)
  //                      c>=512 -> B tile. isB is wave-uniform (no divergence).
  const _Float16* gbase[4];
  unsigned lofs[2][4];
#pragma unroll
  for (int q = 0; q < 4; ++q) {
    int c   = tid + q * 256;
    int isB = c >> 9;
    int cc  = c & 511;
    int row = cc >> 2, seg = (cc & 3) << 3;
    gbase[q] = isB ? (BT + (size_t)(n0 + row) * Kd + seg)
                   : (A  + (size_t)(m0 + row) * Kd + seg);
    _Float16* l0 = (isB ? &Bs[0][0] : &As[0][0]) + row * LDSROW + seg;
    _Float16* l1 = (isB ? &Bs[1][0] : &As[1][0]) + row * LDSROW + seg;
    lofs[0][q] = (unsigned)(size_t)l0;   // low 32 bits of generic ptr = LDS offset
    lofs[1][q] = (unsigned)(size_t)l1;
  }

  fvec8 zero = {0.f,0.f,0.f,0.f,0.f,0.f,0.f,0.f};
  fvec8 acc[2][4];
#pragma unroll
  for (int i = 0; i < 2; ++i)
#pragma unroll
    for (int j = 0; j < 4; ++j) acc[i][j] = zero;

  const int ntiles = Kd / GBK;

  // prologue: stage tiles 0 and 1
#pragma unroll
  for (int q = 0; q < 4; ++q)
    asm volatile("global_load_async_to_lds_b128 %0, %1, off"
                 :: "v"(lofs[0][q]), "v"(gbase[q]) : "memory");
#pragma unroll
  for (int q = 0; q < 4; ++q)
    asm volatile("global_load_async_to_lds_b128 %0, %1, off"
                 :: "v"(lofs[1][q]), "v"(gbase[q] + GBK) : "memory");

  for (int t = 0; t < ntiles; ++t) {
    const int cur = t & 1;
    // wait until tile t's DMAs have landed (per-wave, in-order completion)
    if (t + 1 < ntiles)
      asm volatile("s_wait_asynccnt 0x4" ::: "memory");
    else
      asm volatile("s_wait_asynccnt 0x0" ::: "memory");
    __syncthreads();

    const _Float16* as = &As[cur][0];
    const _Float16* bs = &Bs[cur][0];
    hvec16 af[2], bf[4];
    af[0] = load_fragA(as + (wm * 32 + 0)  * LDSROW, LDSROW, lane);
    af[1] = load_fragA(as + (wm * 32 + 16) * LDSROW, LDSROW, lane);
#pragma unroll
    for (int j = 0; j < 4; ++j)
      bf[j] = load_fragB(bs + (wn * 64 + j * 16) * LDSROW, LDSROW, lane);
#pragma unroll
    for (int i = 0; i < 2; ++i)
#pragma unroll
      for (int j = 0; j < 4; ++j)
        acc[i][j] = __builtin_amdgcn_wmma_f32_16x16x32_f16(
            false, af[i], false, bf[j], (short)0, acc[i][j], false, false);

    __syncthreads();   // everyone finished reading buffer `cur`

    if (t + 2 < ntiles) {
      const int k0 = (t + 2) * GBK;
#pragma unroll
      for (int q = 0; q < 4; ++q)
        asm volatile("global_load_async_to_lds_b128 %0, %1, off"
                     :: "v"(lofs[cur][q]), "v"(gbase[q] + k0) : "memory");
    }
  }

  // epilogue per ISA C/D layout
  const int nlane = lane & 15;
  const int mb    = (lane >> 4) << 3;   // 0 or 8
#pragma unroll
  for (int i = 0; i < 2; ++i)
#pragma unroll
    for (int j = 0; j < 4; ++j) {
      int mrow0 = m0 + wm * 32 + i * 16 + mb;
      int ncol  = n0 + wn * 64 + j * 16 + nlane;
      float bv  = bias ? bias[ncol] : 0.f;
#pragma unroll
      for (int r = 0; r < 8; ++r) {
        float v = acc[i][j][r] + bv;
        if (relu) v = v > 0.f ? v : 0.f;
        size_t off = (size_t)(mrow0 + r) * N + ncol;
        if (C32) C32[off] = v;
        if (C16) C16[off] = (_Float16)v;
      }
    }
}

// ---------------------------------------------------------------------------
// Elementwise / utility kernels
// ---------------------------------------------------------------------------
__global__ void zero_f32_kernel(float* p, long n) {
  long i = (long)blockIdx.x * blockDim.x + threadIdx.x;
  if (i < n) p[i] = 0.f;
}

__global__ void f32_to_f16_kernel(const float* __restrict__ in,
                                  _Float16* __restrict__ out, long n) {
  long i = (long)blockIdx.x * blockDim.x + threadIdx.x;
  if (i < n) out[i] = (_Float16)in[i];
}

// tokens = concat(latents, actions)  -> f32 + f16 copies
__global__ void tokens_kernel(const float* __restrict__ lat,
                              const float* __restrict__ act,
                              float* __restrict__ tok32,
                              _Float16* __restrict__ tok16) {
  long idx = (long)blockIdx.x * blockDim.x + threadIdx.x;
  if (idx >= (long)NTOK * D_) return;
  long t = idx >> 9;          // D_ = 512
  int  d = (int)(idx & 511);
  float v = (d < 256) ? lat[t * 256 + d] : act[t * 256 + (d - 256)];
  tok32[idx] = v;
  tok16[idx] = (_Float16)v;
}

// in[Kd x N] f32 row-major  ->  out[N x Kd] f16  (transpose + convert)
__global__ void transpose_f32_to_f16_kernel(const float* __restrict__ in,
                                            _Float16* __restrict__ out,
                                            int Kd, int N) {
  long idx = (long)blockIdx.x * blockDim.x + threadIdx.x;
  long total = (long)Kd * N;
  if (idx >= total) return;
  int k = (int)(idx % Kd);
  int n = (int)(idx / Kd);
  out[(long)n * Kd + k] = (_Float16)in[(long)k * N + n];
}

// router: logits[b,e] = tok_flat[b,:] . gate_W[:,e]   (one wave per b)
__global__ __launch_bounds__(32) void router_kernel(const float* __restrict__ tok,
                                                    const float* __restrict__ gateW,
                                                    float* __restrict__ logits) {
  int b = blockIdx.x;
  int lane = threadIdx.x;
  const float* t = tok + (size_t)b * FLATD;
  float s0 = 0.f, s1 = 0.f, s2 = 0.f, s3 = 0.f;
  for (int k = lane; k < FLATD; k += 32) {
    float x = t[k];
    const float* w = gateW + (size_t)k * E_;
    s0 += x * w[0]; s1 += x * w[1]; s2 += x * w[2]; s3 += x * w[3];
  }
  for (int m = 16; m; m >>= 1) {
    s0 += __shfl_xor(s0, m, 32); s1 += __shfl_xor(s1, m, 32);
    s2 += __shfl_xor(s2, m, 32); s3 += __shfl_xor(s3, m, 32);
  }
  if (lane == 0) {
    float* o = logits + (size_t)b * E_;
    o[0] = s0; o[1] = s1; o[2] = s2; o[3] = s3;
  }
}

// top-2 softmax gating (jax top_k tie-break: lowest index wins)
__global__ void gates_kernel(const float* __restrict__ logits,
                             float* __restrict__ gates) {
  int b = blockIdx.x * blockDim.x + threadIdx.x;
  if (b >= B_) return;
  float l[E_];
  for (int e = 0; e < E_; ++e) l[e] = logits[b * E_ + e];
  int i0 = 0;
  for (int e = 1; e < E_; ++e) if (l[e] > l[i0]) i0 = e;
  int i1 = -1;
  for (int e = 0; e < E_; ++e) {
    if (e == i0) continue;
    if (i1 < 0 || l[e] > l[i1]) i1 = e;
  }
  float m  = fmaxf(l[i0], l[i1]);
  float e0 = __expf(l[i0] - m), e1 = __expf(l[i1] - m);
  float inv = 1.f / (e0 + e1);
  float g[E_] = {0.f, 0.f, 0.f, 0.f};
  g[i0] = e0 * inv; g[i1] = e1 * inv;
  for (int e = 0; e < E_; ++e) gates[b * E_ + e] = g[e];
}

// deterministic single-block reduction for the load-balancing loss
__global__ __launch_bounds__(256) void lbl_kernel(const float* __restrict__ logits,
                                                  const float* __restrict__ gates,
                                                  float* __restrict__ out_lbl) {
  __shared__ float red[9][256];
  int tid = threadIdx.x;
  float imp[E_] = {0,0,0,0}, cnt[E_] = {0,0,0,0}, lse = 0.f;
  for (int b = tid; b < B_; b += 256) {
    const float* g = gates + (size_t)b * E_;
    for (int e = 0; e < E_; ++e) {
      imp[e] += g[e];
      cnt[e] += (g[e] > 0.f) ? 1.f : 0.f;
    }
    const float* l = logits + (size_t)b * E_;
    float mx = fmaxf(fmaxf(l[0], l[1]), fmaxf(l[2], l[3]));
    float s = __expf(l[0]-mx) + __expf(l[1]-mx) + __expf(l[2]-mx) + __expf(l[3]-mx);
    lse += mx + __logf(s);
  }
  for (int e = 0; e < E_; ++e) { red[e][tid] = imp[e]; red[4+e][tid] = cnt[e]; }
  red[8][tid] = lse;
  __syncthreads();
  for (int s = 128; s; s >>= 1) {
    if (tid < s)
      for (int r = 0; r < 9; ++r) red[r][tid] += red[r][tid + s];
    __syncthreads();
  }
  if (tid == 0) {
    float cv[2];
    for (int grp = 0; grp < 2; ++grp) {
      float mu = 0.f;
      for (int e = 0; e < E_; ++e) mu += red[grp*4 + e][0];
      mu *= 0.25f;
      float var = 0.f;
      for (int e = 0; e < E_; ++e) {
        float d = red[grp*4 + e][0] - mu;
        var += d * d;
      }
      var *= (1.f / 3.f);                    // ddof=1, n=4
      cv[grp] = var / (mu * mu + 1e-10f);
    }
    out_lbl[0] = cv[0] + cv[1] + red[8][0] / (float)B_;
  }
}

// attention: one 64-thread block per batch b. QKV staged in LDS; thread
// (h, i) handles one (head, query-row): 8 scores of len 64, softmax, 64 outs.
__global__ __launch_bounds__(64) void attention_kernel(const float* __restrict__ qkv,
                                                       _Float16* __restrict__ ctx16) {
  __shared__ float sq[NA_ * D_];
  __shared__ float sk[NA_ * D_];
  __shared__ float sv[NA_ * D_];
  int b = blockIdx.x, tid = threadIdx.x;
  const float* base = qkv + (size_t)b * NA_ * (3 * D_);
  for (int idx = tid; idx < NA_ * D_; idx += 64) {
    int t = idx >> 9, d = idx & 511;
    const float* row = base + (size_t)t * (3 * D_);
    sq[idx] = row[d];
    sk[idx] = row[D_ + d];
    sv[idx] = row[2 * D_ + d];
  }
  __syncthreads();
  int h = tid >> 3, i = tid & 7;
  const float* q = sq + i * D_ + h * DH_;
  float sc[NA_];
  float mx = -1e30f;
  for (int j = 0; j < NA_; ++j) {
    const float* k = sk + j * D_ + h * DH_;
    float s = 0.f;
    for (int d = 0; d < DH_; ++d) s += q[d] * k[d];
    s *= 0.125f;                              // 1/sqrt(64)
    sc[j] = s;
    mx = fmaxf(mx, s);
  }
  float sum = 0.f;
  for (int j = 0; j < NA_; ++j) { sc[j] = __expf(sc[j] - mx); sum += sc[j]; }
  float inv = 1.f / sum;
  size_t orow = ((size_t)b * NA_ + i) * D_ + h * DH_;
  for (int d = 0; d < DH_; ++d) {
    float s = 0.f;
    for (int j = 0; j < NA_; ++j) s += sc[j] * sv[j * D_ + h * DH_ + d];
    ctx16[orow + d] = (_Float16)(s * inv);
  }
}

// LayerNorm(res + add) -> x32 (+x16). One wave per token, 8 waves per block.
__global__ __launch_bounds__(256) void ln_kernel(const float* __restrict__ res,
                                                 const float* __restrict__ add,
                                                 const float* __restrict__ gam,
                                                 const float* __restrict__ bet,
                                                 float* __restrict__ x32,
                                                 _Float16* __restrict__ x16) {
  int wave = threadIdx.x >> 5, lane = threadIdx.x & 31;
  size_t t = (size_t)blockIdx.x * 8 + wave;
  const float* r = res + t * D_;
  const float* a = add + t * D_;
  float vals[16];
  float s = 0.f, s2 = 0.f;
  for (int i = 0; i < 16; ++i) {
    float v = r[lane + i * 32] + a[lane + i * 32];
    vals[i] = v; s += v; s2 += v * v;
  }
  for (int m = 16; m; m >>= 1) { s += __shfl_xor(s, m, 32); s2 += __shfl_xor(s2, m, 32); }
  float mu   = s * (1.f / D_);
  float var  = s2 * (1.f / D_) - mu * mu;
  float rstd = rsqrtf(var + LN_EPS);
  for (int i = 0; i < 16; ++i) {
    int d = lane + i * 32;
    float y = (vals[i] - mu) * rstd * gam[d] + bet[d];
    x32[t * D_ + d] = y;
    if (x16) x16[t * D_ + d] = (_Float16)y;
  }
}

// LayerNorm(x + h2), then mix += gate[b,e] * y   (experts run sequentially)
__global__ __launch_bounds__(256) void ln_mix_kernel(const float* __restrict__ x,
                                                     const float* __restrict__ h2,
                                                     const float* __restrict__ gam,
                                                     const float* __restrict__ bet,
                                                     const float* __restrict__ gates,
                                                     int e, float* __restrict__ mix) {
  int wave = threadIdx.x >> 5, lane = threadIdx.x & 31;
  size_t t = (size_t)blockIdx.x * 8 + wave;
  const float* r = x  + t * D_;
  const float* a = h2 + t * D_;
  float vals[16];
  float s = 0.f, s2 = 0.f;
  for (int i = 0; i < 16; ++i) {
    float v = r[lane + i * 32] + a[lane + i * 32];
    vals[i] = v; s += v; s2 += v * v;
  }
  for (int m = 16; m; m >>= 1) { s += __shfl_xor(s, m, 32); s2 += __shfl_xor(s2, m, 32); }
  float mu   = s * (1.f / D_);
  float var  = s2 * (1.f / D_) - mu * mu;
  float rstd = rsqrtf(var + LN_EPS);
  float gate = gates[(t >> 3) * E_ + e];     // batch = token / NA_
  for (int i = 0; i < 16; ++i) {
    int d = lane + i * 32;
    float y = (vals[i] - mu) * rstd * gam[d] + bet[d];
    mix[t * D_ + d] += gate * y;
  }
}

// reward_logits = hid(relu'd) @ head_W2[512x101] + b2
__global__ void head2_kernel(const float* __restrict__ hid,
                             const float* __restrict__ W2,
                             const float* __restrict__ b2,
                             float* __restrict__ out) {
  int idx = blockIdx.x * blockDim.x + threadIdx.x;
  if (idx >= B_ * BINS_) return;
  int b = idx / BINS_, j = idx % BINS_;
  const float* h = hid + (size_t)b * HEAD_;
  float s = b2[j];
  for (int k = 0; k < HEAD_; ++k) s += h[k] * W2[(size_t)k * BINS_ + j];
  out[(size_t)b * BINS_ + j] = s;
}

// ---------------------------------------------------------------------------
// Host launch
// ---------------------------------------------------------------------------
extern "C" void kernel_launch(void* const* d_in, const int* in_sizes, int n_in,
                              void* d_out, int out_size, void* d_ws, size_t ws_size,
                              hipStream_t stream) {
  (void)in_sizes; (void)n_in; (void)out_size; (void)ws_size;

  const float* latents = (const float*)d_in[0];
  const float* actions = (const float*)d_in[1];
  const float* gate_W  = (const float*)d_in[2];
  const float* Wqkv    = (const float*)d_in[3];
  const float* bqkv    = (const float*)d_in[4];
  const float* Wo      = (const float*)d_in[5];
  const float* bo      = (const float*)d_in[6];
  const float* ln1_g   = (const float*)d_in[7];
  const float* ln1_b   = (const float*)d_in[8];
  const float* Wf1     = (const float*)d_in[9];
  const float* bf1     = (const float*)d_in[10];
  const float* Wf2     = (const float*)d_in[11];
  const float* bf2     = (const float*)d_in[12];
  const float* ln2_g   = (const float*)d_in[13];
  const float* ln2_b   = (const float*)d_in[14];
  const float* head_W1 = (const float*)d_in[15];
  const float* head_b1 = (const float*)d_in[16];
  const float* head_W2 = (const float*)d_in[17];
  const float* head_b2 = (const float*)d_in[18];

  float* out_logits = (float*)d_out;                       // [B, BINS]
  float* out_gates  = out_logits + (size_t)B_ * BINS_;     // [B, E]
  float* out_lbl    = out_gates  + (size_t)B_ * E_;        // [1]

  // ---- workspace carve-up (256B aligned) ----
  char* ws = (char*)d_ws;
  size_t off = 0;
  auto alloc = [&](size_t bytes) -> char* {
    char* p = ws + off;
    off += (bytes + 255) & ~(size_t)255;
    return p;
  };
  float*    tok32   = (float*)   alloc((size_t)NTOK * D_ * 4);
  _Float16* tok16   = (_Float16*)alloc((size_t)NTOK * D_ * 2);
  float*    qkv     = (float*)   alloc((size_t)NTOK * 3 * D_ * 4);
  _Float16* ctx16   = (_Float16*)alloc((size_t)NTOK * D_ * 2);
  float*    x32     = (float*)   alloc((size_t)NTOK * D_ * 4);
  _Float16* x16     = (_Float16*)alloc((size_t)NTOK * D_ * 2);
  _Float16* h1_16   = (_Float16*)alloc((size_t)NTOK * FFN_ * 2);
  float*    mix     = (float*)   alloc((size_t)NTOK * D_ * 4);
  float*    hid     = (float*)   alloc((size_t)B_ * HEAD_ * 4);
  float*    logits  = (float*)   alloc((size_t)B_ * E_ * 4);
  _Float16* WqkvT   = (_Float16*)alloc((size_t)E_ * 3 * D_ * D_ * 2);
  _Float16* WoT     = (_Float16*)alloc((size_t)E_ * D_ * D_ * 2);
  _Float16* Wf1T    = (_Float16*)alloc((size_t)E_ * FFN_ * D_ * 2);
  _Float16* Wf2T    = (_Float16*)alloc((size_t)E_ * D_ * FFN_ * 2);
  _Float16* hW1T    = (_Float16*)alloc((size_t)HEAD_ * FLATD * 2);
  // safe aliases into the (dead-by-then) qkv buffer:
  float*    attnout = qkv;                                  // [NTOK, D]
  float*    h2      = qkv + (size_t)NTOK * D_;              // [NTOK, D]
  _Float16* mix16   = tok16;                                // tok16 dead after expert loop

  const long tokN = (long)NTOK * D_;                        // 8388608

  // 1) zero the mixed-output accumulator
  zero_f32_kernel<<<(int)((tokN + 255) / 256), 256, 0, stream>>>(mix, tokN);

  // 2) concat + f32/f16 tokens
  tokens_kernel<<<(int)((tokN + 255) / 256), 256, 0, stream>>>(latents, actions, tok32, tok16);

  // 3) weight transposes ([K x N] f32 -> [N x K] f16)
  auto transpose = [&](const float* in, _Float16* out, int Kd, int N) {
    long total = (long)Kd * N;
    transpose_f32_to_f16_kernel<<<(int)((total + 255) / 256), 256, 0, stream>>>(in, out, Kd, N);
  };
  for (int e = 0; e < E_; ++e) {
    transpose(Wqkv + (size_t)e * D_ * 3 * D_,  WqkvT + (size_t)e * 3 * D_ * D_, D_,   3 * D_);
    transpose(Wo   + (size_t)e * D_ * D_,      WoT   + (size_t)e * D_ * D_,     D_,   D_);
    transpose(Wf1  + (size_t)e * D_ * FFN_,    Wf1T  + (size_t)e * FFN_ * D_,   D_,   FFN_);
    transpose(Wf2  + (size_t)e * FFN_ * D_,    Wf2T  + (size_t)e * D_ * FFN_,   FFN_, D_);
  }
  transpose(head_W1, hW1T, FLATD, HEAD_);

  // 4) router + gating + aux loss
  router_kernel<<<B_, 32, 0, stream>>>(tok32, gate_W, logits);
  gates_kernel<<<(B_ + 255) / 256, 256, 0, stream>>>(logits, out_gates);
  lbl_kernel<<<1, 256, 0, stream>>>(logits, out_gates, out_lbl);

  // 5) experts (sequential; mix += gate * expert_out)
  auto gemm = [&](const _Float16* A, const _Float16* BT, const float* bias,
                  float* C32, _Float16* C16, int M, int N, int Kd, int relu) {
    dim3 grid(M / GBM, N / GBN);
    gemm_f16_wmma_kernel<<<grid, 256, 0, stream>>>(A, BT, bias, C32, C16, M, N, Kd, relu);
  };
  for (int e = 0; e < E_; ++e) {
    // qkv = tok @ Wqkv_e + bqkv_e
    gemm(tok16, WqkvT + (size_t)e * 3 * D_ * D_, bqkv + (size_t)e * 3 * D_,
         qkv, nullptr, NTOK, 3 * D_, D_, 0);
    // attention -> ctx (f16)
    attention_kernel<<<B_, 64, 0, stream>>>(qkv, ctx16);
    // attnout = ctx @ Wo_e + bo_e   (aliased into qkv buffer; qkv is dead)
    gemm(ctx16, WoT + (size_t)e * D_ * D_, bo + (size_t)e * D_,
         attnout, nullptr, NTOK, D_, D_, 0);
    // x = LN1(tok + attnout)
    ln_kernel<<<NTOK / 8, 256, 0, stream>>>(tok32, attnout,
        ln1_g + (size_t)e * D_, ln1_b + (size_t)e * D_, x32, x16);
    // h1 = relu(x @ Wf1_e + bf1_e)  (f16 only)
    gemm(x16, Wf1T + (size_t)e * FFN_ * D_, bf1 + (size_t)e * FFN_,
         nullptr, h1_16, NTOK, FFN_, D_, 1);
    // h2 = h1 @ Wf2_e + bf2_e
    gemm(h1_16, Wf2T + (size_t)e * D_ * FFN_, bf2 + (size_t)e * D_,
         h2, nullptr, NTOK, D_, FFN_, 0);
    // mix += gate_e * LN2(x + h2)
    ln_mix_kernel<<<NTOK / 8, 256, 0, stream>>>(x32, h2,
        ln2_g + (size_t)e * D_, ln2_b + (size_t)e * D_, out_gates, e, mix);
  }

  // 6) reward head
  f32_to_f16_kernel<<<(int)((tokN + 255) / 256), 256, 0, stream>>>(mix, mix16, tokN);
  // hid = relu(mix.reshape(B,4096) @ head_W1 + b1)
  gemm(mix16, hW1T, head_b1, hid, nullptr, B_, HEAD_, FLATD, 1);
  // out = hid @ head_W2 + b2
  head2_kernel<<<(B_ * BINS_ + 255) / 256, 256, 0, stream>>>(hid, head_W2, head_b2, out_logits);
}
